// SimpleMeshProcessor_62440234549300
// MI455X (gfx1250) — matrix operand order repaired and checked by reading, not compiled, not run on hardware
//
#include <hip/hip_runtime.h>
#include <hip/hip_bf16.h>

// GraphNet (15-layer mesh GNN) for MI455X / gfx1250.
// bf16 WMMA (v_wmma_f32_16x16x32_bf16) with fp32 accumulation, fragment-packed
// weights + hidden activations, fused bias/LayerNorm/GELU/residual epilogues,
// and edge aggregation fused as f32 global atomics.

#define NND 10000
#define NE  60000
#define DD  128
#define NL  15

typedef __attribute__((ext_vector_type(16))) __bf16 v16bf;
typedef __attribute__((ext_vector_type(8)))  float  v8f;

__device__ __forceinline__ __bf16 f2bf(float f) {
  union { float f; unsigned u; } x; x.f = f;
  unsigned r = (x.u + 0x7fffu + ((x.u >> 16) & 1u)) >> 16;  // RNE
  union { unsigned short s; __bf16 b; } y; y.s = (unsigned short)r;
  return y.b;
}

// A-matrix (16x32 bf16 per k-tile) packed-fragment index.
// Layout per ISA 7.12.2: lane = h*16 + m (h = K-half), VGPR v holds
// kb = (v<4 ? 2v : 16+2(v-4)) + 8h, low/high halves = kb, kb+1.
// Packed storage: frag[ktile][lane][16 elems].
__device__ __forceinline__ int a_pack_idx(int m, int c) {
  int kt  = c >> 5, kk = c & 31;
  int h   = (kk >> 3) & 1;
  int v   = ((kk >> 4) << 2) | ((kk & 7) >> 1);
  int pos = kk & 1;
  int lane = (h << 4) + m;
  return (((kt << 5) + lane) << 4) + (v << 1) + pos;
}

__device__ __forceinline__ void wave_sum2(float& s, float& s2) {
#pragma unroll
  for (int off = 16; off > 0; off >>= 1) {
    s  += __shfl_xor(s,  off, 32);
    s2 += __shfl_xor(s2, off, 32);
  }
}

// ---------- utility kernels ----------
__global__ void copy_f32(float* __restrict__ d, const float* __restrict__ s, int n) {
  int i = blockIdx.x * 256 + threadIdx.x;
  if (i < n) d[i] = s[i];
}
__global__ void zero_f32(float* __restrict__ p, int n) {
  int i = blockIdx.x * 256 + threadIdx.x;
  if (i < n) p[i] = 0.f;
}
__global__ void count_kernel(float* __restrict__ cnt, const int* __restrict__ col, int n) {
  int i = blockIdx.x * 256 + threadIdx.x;
  if (i < n) atomicAdd(&cnt[col[i]], 1.f);
}
__global__ void rcount_kernel(float* __restrict__ cnt, int n) {
  int i = blockIdx.x * 256 + threadIdx.x;
  if (i < n) cnt[i] = 1.f / fmaxf(cnt[i], 1.f);
}
// Convert f32 weights [L][K][Nw] -> bf16 packed B fragments:
// frag[(kt*(Nw/16)+nt)][lane=h*16+(n&15)][elem=s=2v+pos], kb = (s<8 ? s : 8+s) + 8h (+32kt).
__global__ void pack_b_kernel(__bf16* __restrict__ dst, const float* __restrict__ src,
                              int K, int Nw, int total) {
  int i = blockIdx.x * 256 + threadIdx.x;
  if (i >= total) return;
  int per = K * Nw;
  int l = i / per, rem = i - l * per;
  int k = rem / Nw, n = rem - k * Nw;
  int kt = k >> 5, kk = k & 31;
  int h = (kk >> 3) & 1, q = ((kk >> 4) << 3) | (kk & 7);
  int v = q >> 1, pos = q & 1;
  int lane = (h << 4) + (n & 15);
  int nt = n >> 4;
  size_t di = (size_t)l * per + ((((size_t)kt * (Nw >> 4) + nt) << 5) + lane) * 16 + (v << 1) + pos;
  dst[di] = f2bf(src[i]);
}

// ---------- edge MLP linear1: [16 x 384] @ [384 x 256], LN(256)+GELU -> packed bf16 ----------
__global__ __launch_bounds__(256) void edge_mlp1_kernel(
    const float* __restrict__ xcur, const float* __restrict__ ecur,
    const int* __restrict__ row, const int* __restrict__ col,
    const v16bf* __restrict__ Wp, const float* __restrict__ bias,
    const float* __restrict__ gam, const float* __restrict__ bet,
    __bf16* __restrict__ Hout) {
  __shared__ alignas(32) __bf16 As[12 * 32 * 16];   // 12 k-tiles, packed frags
  __shared__ float Os[16 * 256];
  __shared__ int rs[16], cs[16];
  const int tid = threadIdx.x;
  const int e0 = blockIdx.x << 4;
  if (tid < 16) { rs[tid] = row[e0 + tid]; cs[tid] = col[e0 + tid]; }
  __syncthreads();
  for (int idx = tid; idx < 16 * 128; idx += 256) {
    int r = idx >> 7, c = idx & 127;
    float vx = xcur[(size_t)rs[r] * DD + c];
    float vy = xcur[(size_t)cs[r] * DD + c];
    float ve = ecur[(size_t)(e0 + r) * DD + c];
    As[a_pack_idx(r, c)]       = f2bf(vx);
    As[a_pack_idx(r, c + 128)] = f2bf(vy);
    As[a_pack_idx(r, c + 256)] = f2bf(ve);
  }
  __syncthreads();
  const int wv = tid >> 5, lane = tid & 31;
  const int m = lane & 15, h = lane >> 4;
  const int n0 = (wv << 5) + m;
  const v16bf* Af = (const v16bf*)As;
  v8f acc0 = {}; v8f acc1 = {};
#pragma unroll
  for (int t = 0; t < 12; ++t) {
    v16bf a  = Af[(t << 5) + lane];
    v16bf b0 = Wp[(((t << 4) + (wv << 1)    ) << 5) + lane];
    v16bf b1 = Wp[(((t << 4) + (wv << 1) + 1) << 5) + lane];
    acc0 = __builtin_amdgcn_wmma_f32_16x16x32_bf16(false, a, false, b0, (short)0, acc0, false, false);
    acc1 = __builtin_amdgcn_wmma_f32_16x16x32_bf16(false, a, false, b1, (short)0, acc1, false, false);
  }
#pragma unroll
  for (int j = 0; j < 8; ++j) {
    int rr = j + (h << 3);
    Os[rr * 256 + n0]      = acc0[j];
    Os[rr * 256 + n0 + 16] = acc1[j];
  }
  __syncthreads();
  __bf16* Ho = Hout + (size_t)blockIdx.x * 4096;  // 8 k-tiles * 512 per row-tile
#pragma unroll
  for (int rr = wv << 1; rr < (wv << 1) + 2; ++rr) {
    float s = 0.f, s2 = 0.f;
    for (int c = lane; c < 256; c += 32) {
      float v = Os[rr * 256 + c] + bias[c]; s += v; s2 += v * v;
    }
    wave_sum2(s, s2);
    float mu = s * (1.f / 256.f);
    float rstd = rsqrtf(fmaxf(s2 * (1.f / 256.f) - mu * mu, 0.f) + 1e-5f);
    for (int c = lane; c < 256; c += 32) {
      float v = Os[rr * 256 + c] + bias[c];
      v = (v - mu) * rstd * gam[c] + bet[c];
      v = 0.5f * v * (1.f + erff(v * 0.70710678118654752f));   // exact GELU
      Ho[a_pack_idx(rr, c)] = f2bf(v);
    }
  }
}

// ---------- edge MLP linear2: [16 x 256] @ [256 x 128], LN(128)+residual, fused atomic agg ----------
__global__ __launch_bounds__(256) void edge_mlp2_kernel(
    const v16bf* __restrict__ Hp, const int* __restrict__ col,
    const v16bf* __restrict__ Wp, const float* __restrict__ bias,
    const float* __restrict__ gam, const float* __restrict__ bet,
    float* __restrict__ ecur, float* __restrict__ agg) {
  __shared__ float Os[16 * 128];
  const int tid = threadIdx.x;
  const int e0 = blockIdx.x << 4;
  const int wv = tid >> 5, lane = tid & 31;
  const int m = lane & 15, h = lane >> 4;
  const v16bf* Af = Hp + (size_t)blockIdx.x * 256;  // 8 k-tiles * 32 lane-frags
  v8f acc = {};
#pragma unroll
  for (int t = 0; t < 8; ++t) {
    v16bf a = Af[(t << 5) + lane];
    v16bf b = Wp[(((t << 3) + wv) << 5) + lane];
    acc = __builtin_amdgcn_wmma_f32_16x16x32_bf16(false, a, false, b, (short)0, acc, false, false);
  }
  const int n0 = (wv << 4) + m;
#pragma unroll
  for (int j = 0; j < 8; ++j) Os[(j + (h << 3)) * 128 + n0] = acc[j];
  __syncthreads();
#pragma unroll
  for (int rr = wv << 1; rr < (wv << 1) + 2; ++rr) {
    float s = 0.f, s2 = 0.f;
    for (int c = lane; c < 128; c += 32) {
      float v = Os[rr * 128 + c] + bias[c]; s += v; s2 += v * v;
    }
    wave_sum2(s, s2);
    float mu = s * (1.f / 128.f);
    float rstd = rsqrtf(fmaxf(s2 * (1.f / 128.f) - mu * mu, 0.f) + 1e-5f);
    int e = e0 + rr;
    int cd = col[e];
    for (int c = lane; c < 128; c += 32) {
      float v = Os[rr * 128 + c] + bias[c];
      v = (v - mu) * rstd * gam[c] + bet[c];
      size_t off = (size_t)e * DD + c;
      float nv = ecur[off] + v;
      ecur[off] = nv;
      atomicAdd(&agg[(size_t)cd * DD + c], nv);
    }
  }
}

// ---------- node MLP linear1: [16 x 256] @ [256 x 256], LN(256)+GELU -> packed bf16 ----------
__global__ __launch_bounds__(256) void node_mlp1_kernel(
    const float* __restrict__ xcur, const float* __restrict__ agg,
    const float* __restrict__ rcount,
    const v16bf* __restrict__ Wp, const float* __restrict__ bias,
    const float* __restrict__ gam, const float* __restrict__ bet,
    __bf16* __restrict__ Hout) {
  __shared__ alignas(32) __bf16 As[8 * 32 * 16];
  __shared__ float Os[16 * 256];
  const int tid = threadIdx.x;
  const int n0r = blockIdx.x << 4;
  for (int idx = tid; idx < 16 * 128; idx += 256) {
    int r = idx >> 7, c = idx & 127;
    int nn = n0r + r;
    As[a_pack_idx(r, c)]       = f2bf(xcur[(size_t)nn * DD + c]);
    As[a_pack_idx(r, c + 128)] = f2bf(agg[(size_t)nn * DD + c] * rcount[nn]);
  }
  __syncthreads();
  const int wv = tid >> 5, lane = tid & 31;
  const int m = lane & 15, h = lane >> 4;
  const int n0 = (wv << 5) + m;
  const v16bf* Af = (const v16bf*)As;
  v8f acc0 = {}; v8f acc1 = {};
#pragma unroll
  for (int t = 0; t < 8; ++t) {
    v16bf a  = Af[(t << 5) + lane];
    v16bf b0 = Wp[(((t << 4) + (wv << 1)    ) << 5) + lane];
    v16bf b1 = Wp[(((t << 4) + (wv << 1) + 1) << 5) + lane];
    acc0 = __builtin_amdgcn_wmma_f32_16x16x32_bf16(false, a, false, b0, (short)0, acc0, false, false);
    acc1 = __builtin_amdgcn_wmma_f32_16x16x32_bf16(false, a, false, b1, (short)0, acc1, false, false);
  }
#pragma unroll
  for (int j = 0; j < 8; ++j) {
    int rr = j + (h << 3);
    Os[rr * 256 + n0]      = acc0[j];
    Os[rr * 256 + n0 + 16] = acc1[j];
  }
  __syncthreads();
  __bf16* Ho = Hout + (size_t)blockIdx.x * 4096;
#pragma unroll
  for (int rr = wv << 1; rr < (wv << 1) + 2; ++rr) {
    float s = 0.f, s2 = 0.f;
    for (int c = lane; c < 256; c += 32) {
      float v = Os[rr * 256 + c] + bias[c]; s += v; s2 += v * v;
    }
    wave_sum2(s, s2);
    float mu = s * (1.f / 256.f);
    float rstd = rsqrtf(fmaxf(s2 * (1.f / 256.f) - mu * mu, 0.f) + 1e-5f);
    for (int c = lane; c < 256; c += 32) {
      float v = Os[rr * 256 + c] + bias[c];
      v = (v - mu) * rstd * gam[c] + bet[c];
      v = 0.5f * v * (1.f + erff(v * 0.70710678118654752f));
      Ho[a_pack_idx(rr, c)] = f2bf(v);
    }
  }
}

// ---------- node MLP linear2: [16 x 256] @ [256 x 128], LN(128)+residual into x ----------
__global__ __launch_bounds__(256) void node_mlp2_kernel(
    const v16bf* __restrict__ Hp,
    const v16bf* __restrict__ Wp, const float* __restrict__ bias,
    const float* __restrict__ gam, const float* __restrict__ bet,
    float* __restrict__ xcur) {
  __shared__ float Os[16 * 128];
  const int tid = threadIdx.x;
  const int n0r = blockIdx.x << 4;
  const int wv = tid >> 5, lane = tid & 31;
  const int m = lane & 15, h = lane >> 4;
  const v16bf* Af = Hp + (size_t)blockIdx.x * 256;
  v8f acc = {};
#pragma unroll
  for (int t = 0; t < 8; ++t) {
    v16bf a = Af[(t << 5) + lane];
    v16bf b = Wp[(((t << 3) + wv) << 5) + lane];
    acc = __builtin_amdgcn_wmma_f32_16x16x32_bf16(false, a, false, b, (short)0, acc, false, false);
  }
  const int n0 = (wv << 4) + m;
#pragma unroll
  for (int j = 0; j < 8; ++j) Os[(j + (h << 3)) * 128 + n0] = acc[j];
  __syncthreads();
#pragma unroll
  for (int rr = wv << 1; rr < (wv << 1) + 2; ++rr) {
    float s = 0.f, s2 = 0.f;
    for (int c = lane; c < 128; c += 32) {
      float v = Os[rr * 128 + c] + bias[c]; s += v; s2 += v * v;
    }
    wave_sum2(s, s2);
    float mu = s * (1.f / 128.f);
    float rstd = rsqrtf(fmaxf(s2 * (1.f / 128.f) - mu * mu, 0.f) + 1e-5f);
    int nn = n0r + rr;
    for (int c = lane; c < 128; c += 32) {
      float v = Os[rr * 128 + c] + bias[c];
      v = (v - mu) * rstd * gam[c] + bet[c];
      xcur[(size_t)nn * DD + c] += v;
    }
  }
}

extern "C" void kernel_launch(void* const* d_in, const int* in_sizes, int n_in,
                              void* d_out, int out_size, void* d_ws, size_t ws_size,
                              hipStream_t stream) {
  (void)in_sizes; (void)n_in; (void)out_size; (void)ws_size;
  const float* x     = (const float*)d_in[0];
  const int*   eidx  = (const int*)d_in[1];
  const float* eattr = (const float*)d_in[2];
  const float* ew1 = (const float*)d_in[3];
  const float* eb1 = (const float*)d_in[4];
  const float* eg1 = (const float*)d_in[5];
  const float* ebt1= (const float*)d_in[6];
  const float* ew2 = (const float*)d_in[7];
  const float* eb2 = (const float*)d_in[8];
  const float* eg2 = (const float*)d_in[9];
  const float* ebt2= (const float*)d_in[10];
  const float* nw1 = (const float*)d_in[11];
  const float* nb1 = (const float*)d_in[12];
  const float* ng1 = (const float*)d_in[13];
  const float* nbt1= (const float*)d_in[14];
  const float* nw2 = (const float*)d_in[15];
  const float* nb2 = (const float*)d_in[16];
  const float* ng2 = (const float*)d_in[17];
  const float* nbt2= (const float*)d_in[18];

  const int* row = eidx;
  const int* col = eidx + NE;

  float* xcur = (float*)d_out;                 // [N, D] (output 0)
  float* ecur = xcur + (size_t)NND * DD;       // [E, D] (output 1)

  size_t off = 0;
  char* wsb = (char*)d_ws;
  auto take = [&](size_t bytes) -> void* {
    void* p = wsb + off;
    off += (bytes + 255) & ~(size_t)255;
    return p;
  };
  float*  rcount = (float*)take((size_t)NND * 4);
  __bf16* ew1b = (__bf16*)take((size_t)NL * 384 * 256 * 2);
  __bf16* ew2b = (__bf16*)take((size_t)NL * 256 * 128 * 2);
  __bf16* nw1b = (__bf16*)take((size_t)NL * 256 * 256 * 2);
  __bf16* nw2b = (__bf16*)take((size_t)NL * 256 * 128 * 2);
  __bf16* hedge = (__bf16*)take((size_t)NE * 256 * 2);
  __bf16* hnode = (__bf16*)take((size_t)NND * 256 * 2);
  float*  aggp  = (float*)take((size_t)NND * DD * 4);

  // init outputs + neighbor counts
  copy_f32<<<(NND * DD + 255) / 256, 256, 0, stream>>>(xcur, x, NND * DD);
  copy_f32<<<(NE * DD + 255) / 256, 256, 0, stream>>>(ecur, eattr, NE * DD);
  zero_f32<<<(NND + 255) / 256, 256, 0, stream>>>(rcount, NND);
  count_kernel<<<(NE + 255) / 256, 256, 0, stream>>>(rcount, col, NE);
  rcount_kernel<<<(NND + 255) / 256, 256, 0, stream>>>(rcount, NND);

  // pack all weights to bf16 B-fragments
  {
    int t1 = NL * 384 * 256, t2 = NL * 256 * 128, t3 = NL * 256 * 256;
    pack_b_kernel<<<(t1 + 255) / 256, 256, 0, stream>>>(ew1b, ew1, 384, 256, t1);
    pack_b_kernel<<<(t2 + 255) / 256, 256, 0, stream>>>(ew2b, ew2, 256, 128, t2);
    pack_b_kernel<<<(t3 + 255) / 256, 256, 0, stream>>>(nw1b, nw1, 256, 256, t3);
    pack_b_kernel<<<(t2 + 255) / 256, 256, 0, stream>>>(nw2b, nw2, 256, 128, t2);
  }

  for (int i = 0; i < NL; ++i) {
    zero_f32<<<(NND * DD + 255) / 256, 256, 0, stream>>>(aggp, NND * DD);
    edge_mlp1_kernel<<<NE / 16, 256, 0, stream>>>(
        xcur, ecur, row, col,
        (const v16bf*)(ew1b + (size_t)i * 384 * 256),
        eb1 + (size_t)i * 256, eg1 + (size_t)i * 256, ebt1 + (size_t)i * 256, hedge);
    edge_mlp2_kernel<<<NE / 16, 256, 0, stream>>>(
        (const v16bf*)hedge, col,
        (const v16bf*)(ew2b + (size_t)i * 256 * 128),
        eb2 + (size_t)i * 128, eg2 + (size_t)i * 128, ebt2 + (size_t)i * 128, ecur, aggp);
    node_mlp1_kernel<<<NND / 16, 256, 0, stream>>>(
        xcur, aggp, rcount,
        (const v16bf*)(nw1b + (size_t)i * 256 * 256),
        nb1 + (size_t)i * 256, ng1 + (size_t)i * 256, nbt1 + (size_t)i * 256, hnode);
    node_mlp2_kernel<<<NND / 16, 256, 0, stream>>>(
        (const v16bf*)hnode,
        (const v16bf*)(nw2b + (size_t)i * 256 * 128),
        nb2 + (size_t)i * 128, ng2 + (size_t)i * 128, nbt2 + (size_t)i * 128, xcur);
  }
}
